// GraphVertModel_23914377904734
// MI455X (gfx1250) — matrix operand order repaired and checked by reading, not compiled, not run on hardware
//
#include <hip/hip_runtime.h>

// Problem constants (match reference)
#define BB 32
#define MM 512
#define FN 64
#define DD 128
#define GSZ 4
#define LL 8

#if defined(__AMDGCN__) && __has_builtin(__builtin_amdgcn_global_load_async_to_lds_b128) && \
    __has_builtin(__builtin_amdgcn_s_wait_asynccnt)
#define USE_ASYNC_LDS 1
#else
#define USE_ASYNC_LDS 0
#endif

typedef __attribute__((ext_vector_type(16))) __bf16 v16bf;
typedef __attribute__((ext_vector_type(8)))  float  v8f;

#if USE_ASYNC_LDS
// The async-to-LDS builtin takes GCC-vector int4 pointers (global src, LDS dst).
typedef int v4i_vs __attribute__((vector_size(16)));
typedef __attribute__((address_space(1))) v4i_vs* gv4ip;
typedef __attribute__((address_space(3))) v4i_vs* lv4ip;
#endif

union Frag16 {
    v16bf v;
    uint4 u4[2];
    unsigned short s[16];
};

__device__ __forceinline__ unsigned short f2bf(float f) {
    unsigned int u = __builtin_bit_cast(unsigned int, f);
    unsigned int r = u + 0x7FFFu + ((u >> 16) & 1u);   // round-to-nearest-even
    return (unsigned short)(r >> 16);
}

__device__ __forceinline__ v8f wmma_bf16(v16bf a, v16bf b, v8f c) {
    // D = A(16x32 bf16) x B(32x16 bf16) + C(16x16 f32)
    return __builtin_amdgcn_wmma_f32_16x16x32_bf16(
        /*neg_a=*/false, a, /*neg_b=*/false, b,
        /*c_mod=*/(short)0, c, /*reuse_a=*/false, /*reuse_b=*/false);
}

// ---------------------------------------------------------------------------
// fp32 -> bf16 bulk convert (n4 = n/4, n divisible by 4)
// ---------------------------------------------------------------------------
__global__ void cvt_f32_bf16_kernel(const float* __restrict__ src,
                                    unsigned short* __restrict__ dst, int n4) {
    int i = blockIdx.x * blockDim.x + threadIdx.x;
    if (i >= n4) return;
    float4 f = ((const float4*)src)[i];
    ushort4 o;
    o.x = f2bf(f.x); o.y = f2bf(f.y); o.z = f2bf(f.z); o.w = f2bf(f.w);
    ((ushort4*)dst)[i] = o;
}

// ---------------------------------------------------------------------------
// Pack weights [nmat][C][128] f32 -> B-fragment layout [nmat][C/32][8][32][16] bf16
// ---------------------------------------------------------------------------
__global__ void pack_w_kernel(const float* __restrict__ src,
                              unsigned short* __restrict__ dst,
                              int C, int nmat) {
    int fid  = blockIdx.x * (blockDim.x >> 5) + (threadIdx.x >> 5);
    int lane = threadIdx.x & 31;
    int ktiles = C >> 5;
    int nfrag  = nmat * ktiles * 8;
    if (fid >= nfrag) return;
    int nt  = fid & 7;
    int kt  = (fid >> 3) % ktiles;
    int mat = fid / (8 * ktiles);
    int p  = nt * 16 + (lane & 15);
    int k0 = kt * 32 + ((lane & 16) ? 16 : 0);
    const float* s = src + (long)mat * C * DD;
    Frag16 t;
    #pragma unroll
    for (int j = 0; j < 16; ++j) t.s[j] = f2bf(s[(long)(k0 + j) * DD + p]);
    unsigned short* d = dst + ((long)fid * 32 + lane) * 16;
    *(uint4*)d       = t.u4[0];
    *(uint4*)(d + 8) = t.u4[1];
}

// ---------------------------------------------------------------------------
// Step A: multi[b][g][m][p] = h[b][m][:] @ W[g][:][p] + bias[g][p]  (bf16 WMMA)
// Output written in packed B-fragment layout: [b][g][m/32][8][32][16] bf16
// ---------------------------------------------------------------------------
template <int C>
__global__ __launch_bounds__(256, 1)
void linear_kernel(const unsigned short* __restrict__ h,      // [B][M][C] bf16
                   const unsigned short* __restrict__ wpack,  // [GS][C/32][8][32][16]
                   const float* __restrict__ bias,            // [GS][128]
                   unsigned short* __restrict__ multi) {      // packed out
    __shared__ unsigned short ldsT[8][DD][32];   // per-wave transpose area [p][atom], 64 KB

    int wave = threadIdx.x >> 5;
    int lane = threadIdx.x & 31;
    int l15  = lane & 15;
    int hiK  = lane >> 4;            // 0 | 1

    int blk   = blockIdx.x;
    int mhalf = blk & 1;
    int g     = (blk >> 1) & 3;
    int b     = blk >> 3;
    int atom0 = mhalf * 256 + wave * 32;

    const unsigned short* hb = h + ((long)b * MM + atom0) * C;
    const unsigned short* wg = wpack + (long)g * (C >> 5) * 8 * 32 * 16;

    v8f acc[2][8] = {};

    #pragma unroll
    for (int kt = 0; kt < (C >> 5); ++kt) {
        Frag16 afr[2];
        #pragma unroll
        for (int r = 0; r < 2; ++r) {
            const unsigned short* arow = hb + (long)(r * 16 + l15) * C + kt * 32 + hiK * 16;
            afr[r].u4[0] = *(const uint4*)arow;          // K +0..7   (or +16..23)
            afr[r].u4[1] = *(const uint4*)(arow + 8);    // K +8..15  (or +24..31)
        }
        #pragma unroll
        for (int nt = 0; nt < 8; ++nt) {
            const unsigned short* bp = wg + (((long)kt * 8 + nt) * 32 + lane) * 16;
            Frag16 bfr;
            bfr.u4[0] = *(const uint4*)bp;
            bfr.u4[1] = *(const uint4*)(bp + 8);
            acc[0][nt] = wmma_bf16(afr[0].v, bfr.v, acc[0][nt]);
            acc[1][nt] = wmma_bf16(afr[1].v, bfr.v, acc[1][nt]);
        }
    }

    // bias add, f32->bf16, transpose through LDS into B-fragment layout
    #pragma unroll
    for (int r = 0; r < 2; ++r) {
        #pragma unroll
        for (int nt = 0; nt < 8; ++nt) {
            int p = nt * 16 + l15;
            float bv = bias[g * DD + p];
            v8f a = acc[r][nt];
            unsigned short t8[8];
            #pragma unroll
            for (int j = 0; j < 8; ++j) t8[j] = f2bf(a[j] + bv);
            *(uint4*)&ldsT[wave][p][r * 16 + hiK * 8] = *(const uint4*)t8;
        }
    }
    __syncthreads();

    int ktB = atom0 >> 5;  // this wave's 32 atoms == K-tile index of the next GEMM
    unsigned short* mp = multi + ((((long)b * GSZ + g) * 16 + ktB) * 8) * 32 * 16;
    #pragma unroll
    for (int nt = 0; nt < 8; ++nt) {
        int p = nt * 16 + l15;
        const unsigned short* sT = &ldsT[wave][p][hiK * 16];
        uint4 q0 = *(const uint4*)sT;
        uint4 q1 = *(const uint4*)(sT + 8);
        unsigned short* dp = mp + ((long)nt * 32 + lane) * 16;
        *(uint4*)dp       = q0;
        *(uint4*)(dp + 8) = q1;
    }
}

// ---------------------------------------------------------------------------
// Step B+C: xout = G @ multi ; h' = max_g relu(xout) (+ residual)
// grid = B*(M/128), block = 256 (8 waves); wave owns 16 atoms x 128 cols.
// Staging: double-buffered GLOBAL_LOAD_ASYNC_TO_LDS_B128 (ASYNCcnt) when
// available; next B-tile DMA overlaps the current tile's WMMAs.
// ---------------------------------------------------------------------------
__global__ __launch_bounds__(256, 1)
void adj_kernel(const unsigned short* __restrict__ G,      // [B][GS][M][M] bf16
                const unsigned short* __restrict__ multi,  // packed [B][GS][16][8][32][16]
                const float* __restrict__ hres,            // residual (f32) or ignored
                float* __restrict__ houtf,                 // [B][M][D] f32
                unsigned short* __restrict__ houtb,        // [B][M][D] bf16
                int use_res) {
    __shared__ unsigned short btile[2][32 * 128];   // two 8 KB K-tile buffers

    int t    = threadIdx.x;
    int lane = t & 31;
    int wave = t >> 5;
    int l15  = lane & 15;
    int hiK  = lane >> 4;

    int b    = blockIdx.x >> 2;
    int mblk = blockIdx.x & 3;
    int m0   = mblk * 128 + wave * 16;

    v8f hmax[8] = {};   // relu makes everything >= 0, so 0-init == true channel max

    for (int g = 0; g < GSZ; ++g) {
        const unsigned short* Gbg = G + (((long)b * GSZ + g) * MM + m0) * MM;
        const unsigned short* mpg = multi + ((long)b * GSZ + g) * (16 * 8 * 32 * 16);

        v8f acc[8] = {};

#if USE_ASYNC_LDS
#define STAGE_ASYNC(KT, PB)                                                          \
    do {                                                                             \
        const unsigned short* s_ = mpg + (long)(KT) * 4096;                          \
        __builtin_amdgcn_global_load_async_to_lds_b128(                              \
            (gv4ip)(s_ + (size_t)t * 8),                                             \
            (lv4ip)(&btile[PB][t * 8]), 0, 0);                                       \
        __builtin_amdgcn_global_load_async_to_lds_b128(                              \
            (gv4ip)(s_ + (size_t)(t + 256) * 8),                                     \
            (lv4ip)(&btile[PB][(t + 256) * 8]), 0, 0);                               \
    } while (0)
        STAGE_ASYNC(0, 0);   // prologue: tile 0 -> buffer 0 (barriers below cover reuse)
#endif

        #pragma unroll
        for (int kt = 0; kt < 16; ++kt) {
            const int pb = kt & 1;
#if USE_ASYNC_LDS
            __syncthreads();                       // (A) nobody still reads btile[pb^1]
            if (kt + 1 < 16) {
                STAGE_ASYNC(kt + 1, pb ^ 1);
                __builtin_amdgcn_s_wait_asynccnt(2);  // my slice of btile[pb] landed
            } else {
                __builtin_amdgcn_s_wait_asynccnt(0);
            }
            __syncthreads();                       // (B) everyone's slice landed
            const unsigned short* bbase = btile[pb];
#else
            __syncthreads();
            {   // synchronous stage: 4096 bf16 = 512 uint4, 2 per thread
                const uint4* s = (const uint4*)(mpg + (long)kt * 4096);
                uint4* d = (uint4*)btile[0];
                d[t]       = s[t];
                d[t + 256] = s[t + 256];
            }
            __syncthreads();
            const unsigned short* bbase = btile[0];
            (void)pb;
#endif
            // A fragment: 16x32 bf16 tile of G rows m0..m0+15, K = kt*32..+31
            const unsigned short* arow = Gbg + (long)l15 * MM + kt * 32 + hiK * 16;
            Frag16 af;
            af.u4[0] = *(const uint4*)arow;
            af.u4[1] = *(const uint4*)(arow + 8);

            // software-pipelined B fragments: load nt+1 before WMMA nt
            Frag16 bcur, bnxt;
            bcur.u4[0] = *(const uint4*)(bbase + (size_t)lane * 16);
            bcur.u4[1] = *(const uint4*)(bbase + (size_t)lane * 16 + 8);
            #pragma unroll
            for (int nt = 0; nt < 8; ++nt) {
                if (nt < 7) {
                    const unsigned short* bp = bbase + ((size_t)(nt + 1) * 32 + lane) * 16;
                    bnxt.u4[0] = *(const uint4*)bp;
                    bnxt.u4[1] = *(const uint4*)(bp + 8);
                }
                acc[nt] = wmma_bf16(af.v, bcur.v, acc[nt]);
                if (nt < 7) bcur = bnxt;
            }
        }

        // fused ReLU + channel max
        #pragma unroll
        for (int nt = 0; nt < 8; ++nt) {
            #pragma unroll
            for (int j = 0; j < 8; ++j) {
                float v = acc[nt][j];
                v = v > 0.0f ? v : 0.0f;
                hmax[nt][j] = v > hmax[nt][j] ? v : hmax[nt][j];
            }
        }
    }

    // epilogue: residual add, dual write f32 + bf16
    #pragma unroll
    for (int nt = 0; nt < 8; ++nt) {
        int p = nt * 16 + l15;
        #pragma unroll
        for (int j = 0; j < 8; ++j) {
            int m = m0 + hiK * 8 + j;
            long idx = ((long)b * MM + m) * DD + p;
            float v = hmax[nt][j];
            if (use_res) v += hres[idx];
            houtf[idx] = v;
            houtb[idx] = f2bf(v);
        }
    }
}

// ---------------------------------------------------------------------------
// Final: out[b][m] = h[b][m][:] . Wout + bout   (f32)
// ---------------------------------------------------------------------------
__global__ void out_kernel(const float* __restrict__ hf,
                           const float* __restrict__ Wout,
                           const float* __restrict__ bout,
                           float* __restrict__ out) {
    int i = blockIdx.x * blockDim.x + threadIdx.x;
    if (i >= BB * MM) return;
    const float4* hp = (const float4*)(hf + (long)i * DD);
    const float4* wp = (const float4*)Wout;
    float s = 0.0f;
    #pragma unroll
    for (int j = 0; j < DD / 4; ++j) {
        float4 a = hp[j], w = wp[j];
        s += a.x * w.x + a.y * w.y + a.z * w.z + a.w * w.w;
    }
    out[i] = s + bout[0];
}

// ---------------------------------------------------------------------------
extern "C" void kernel_launch(void* const* d_in, const int* in_sizes, int n_in,
                              void* d_out, int out_size, void* d_ws, size_t ws_size,
                              hipStream_t stream) {
    const float* G    = (const float*)d_in[0];   // [B][GS][M][M]
    const float* x    = (const float*)d_in[1];   // [B][M][FN]
    const float* W0   = (const float*)d_in[2];   // [GS][FN][D]
    const float* b0   = (const float*)d_in[3];   // [GS][D]
    const float* Wl   = (const float*)d_in[4];   // [L-1][GS][D][D]
    const float* bl   = (const float*)d_in[5];   // [L-1][GS][D]
    const float* Wout = (const float*)d_in[6];   // [D][1]
    const float* bout = (const float*)d_in[7];   // [1]
    float* out = (float*)d_out;                  // [B][M][1] f32

    // workspace carve-up
    char* ws = (char*)d_ws;
    const size_t nG  = (size_t)BB * GSZ * MM * MM;        // 33,554,432
    const size_t nX  = (size_t)BB * MM * FN;              //  1,048,576
    const size_t nH  = (size_t)BB * MM * DD;              //  2,097,152
    const size_t nMu = (size_t)BB * GSZ * MM * DD;        //  8,388,608
    const size_t nW0 = (size_t)GSZ * FN * DD;             //     32,768
    const size_t nWl = (size_t)(LL - 1) * GSZ * DD * DD;  //    458,752

    unsigned short* Gb  = (unsigned short*)ws;                 ws += nG  * 2;
    unsigned short* xb  = (unsigned short*)ws;                 ws += nX  * 2;
    unsigned short* hb  = (unsigned short*)ws;                 ws += nH  * 2;
    float*          hf  = (float*)ws;                         ws += nH  * 4;
    unsigned short* mp  = (unsigned short*)ws;                 ws += nMu * 2;
    unsigned short* w0p = (unsigned short*)ws;                 ws += nW0 * 2;
    unsigned short* wlp = (unsigned short*)ws;                 ws += nWl * 2;
    (void)ws_size; (void)in_sizes; (void)n_in; (void)out_size;

    // 1) converts + weight packing
    cvt_f32_bf16_kernel<<<(int)(nG / 4 / 256), 256, 0, stream>>>(G, Gb, (int)(nG / 4));
    cvt_f32_bf16_kernel<<<(int)(nX / 4 / 256), 256, 0, stream>>>(x, xb, (int)(nX / 4));
    pack_w_kernel<<<(64 + 7) / 8, 256, 0, stream>>>(W0, w0p, FN, GSZ);
    pack_w_kernel<<<(896 + 7) / 8, 256, 0, stream>>>(Wl, wlp, DD, (LL - 1) * GSZ);

    const int linGrid = BB * GSZ * (MM / 256);  // 256
    const int adjGrid = BB * (MM / 128);        // 128

    // 2) layer 0 (C = FN, no residual)
    linear_kernel<FN><<<linGrid, 256, 0, stream>>>(xb, w0p, b0, mp);
    adj_kernel<<<adjGrid, 256, 0, stream>>>(Gb, mp, hf, hf, hb, 0);

    // 3) layers 1..L-1 (C = D, residual)
    for (int i = 0; i < LL - 1; ++i) {
        const unsigned short* wlay = wlp + (size_t)i * GSZ * DD * DD;
        const float* blay = bl + (size_t)i * GSZ * DD;
        linear_kernel<DD><<<linGrid, 256, 0, stream>>>(hb, wlay, blay, mp);
        adj_kernel<<<adjGrid, 256, 0, stream>>>(Gb, mp, hf, hf, hb, 1);
    }

    // 4) output projection
    out_kernel<<<(BB * MM) / 256, 256, 0, stream>>>(hf, Wout, bout, out);
}
